// RNN_395136991617
// MI455X (gfx1250) — compile-verified, matrix operand order checked by reading, not run on hardware
//
#include <hip/hip_runtime.h>
#include <hip/hip_bf16.h>

// ---------------------------------------------------------------------------
// 2-layer LSTM (B=256, T=512, I=128, H=256) + FC(1000) on gfx1250 (CDNA5).
//   * f16 operands, f32 WMMA accumulate (v_wmma_f32_16x16x32_f16).
//   * Fused input projection: gates[16,1024] = [x_t | h] @ [Wih|Whh]^T,
//     K = I+H (layer0: 384, layer1: 512). Weights stay L2-resident.
//   * One persistent kernel, 32 workgroups x 512 threads:
//       blocks  0..15: layer 0, batch tile 16 rows each
//       blocks 16..31: layer 1, pipelined one timestep behind layer 0 via
//                      per-tile release/acquire flags (agent scope).
//   * 16 waves/WG; wave w owns hidden slice [16w,16w+16): all 4 gates
//     wave-local, cell state c held in VGPRs for all 512 steps.
//   * Per-timestep asm memory clobber blocks LICM from hoisting+spilling the
//     weight fragments to scratch (observed in round 2) -> weights are
//     re-read from L2 each step as intended.
//   * h tile in LDS (f16, padded stride 264 halves -> conflict-free
//     ds_load_b128 A-fragments), two barriers per step.
//   * FC: WMMA GEMM over 16x63 tiles, zero-padded f16 weights.
// ---------------------------------------------------------------------------

typedef __attribute__((ext_vector_type(16))) _Float16 v16h;
typedef __attribute__((ext_vector_type(8)))  _Float16 v8h;
typedef __attribute__((ext_vector_type(8)))  float    v8f;

union V16U { v16h v; v8h h[2]; };

#define B_  256
#define T_  512
#define I_  128
#define H_  256
#define C_  1000
#define G_  1024   // 4*H
#define CPAD_ 1008 // C rounded up to 16

// ---- fragment loader: wave32 16-bit A/B layout (16x32 / 32x16) ------------
// lane L: row = row0 + (L&15); halves [0..7] at k0 + ((L>>4)*8),
//         halves [8..15] at +16.  Two b128 loads.
__device__ __forceinline__ v16h load_frag(const _Float16* __restrict__ base,
                                          int row0, int stride, int k0, int lane) {
    const int row = row0 + (lane & 15);
    const int k   = k0 + ((lane >> 4) << 3);
    const _Float16* p = base + (size_t)row * (size_t)stride + k;
    V16U u;
    u.h[0] = *(const v8h*)(p);
    u.h[1] = *(const v8h*)(p + 16);
    return u.v;
}

__device__ __forceinline__ float sigf(float x) {
    return 1.0f / (1.0f + __expf(-x));
}

// ---------------------------------------------------------------------------
// Prep kernels
// ---------------------------------------------------------------------------
__global__ void cvt_f32_to_f16(const float* __restrict__ src,
                               _Float16* __restrict__ dst, int n) {
    int i = blockIdx.x * blockDim.x + threadIdx.x;
    if (i < n) dst[i] = (_Float16)src[i];
}

// Wcat[g, 0:kin] = Wih[g,:] ; Wcat[g, kin:kin+H] = Whh[g,:]
__global__ void build_wcat(const float* __restrict__ Wih,
                           const float* __restrict__ Whh,
                           _Float16* __restrict__ Wcat, int kin) {
    const int kcat = kin + H_;
    int i = blockIdx.x * blockDim.x + threadIdx.x;
    if (i >= G_ * kcat) return;
    int g = i / kcat, k = i - g * kcat;
    float v = (k < kin) ? Wih[g * kin + k] : Whh[g * H_ + (k - kin)];
    Wcat[i] = (_Float16)v;
}

__global__ void build_bsum(const float* __restrict__ a,
                           const float* __restrict__ b,
                           float* __restrict__ s) {
    int i = blockIdx.x * blockDim.x + threadIdx.x;
    if (i < G_) s[i] = a[i] + b[i];
}

__global__ void build_fcw(const float* __restrict__ fcW,
                          _Float16* __restrict__ fcWp) {
    int i = blockIdx.x * blockDim.x + threadIdx.x;
    if (i >= CPAD_ * H_) return;
    int r = i / H_, k = i - r * H_;
    fcWp[i] = (_Float16)((r < C_) ? fcW[r * H_ + k] : 0.0f);
}

__global__ void zero_flags(int* __restrict__ flags, int n) {
    int i = threadIdx.x;
    if (i < n) flags[i] = 0;
}

// ---------------------------------------------------------------------------
// One LSTM layer body, persistent over time. 512 threads = 16 waves; wave w
// owns hidden slice [16w, 16w+16). waitFlag: producer progress (steps done)
// for this batch tile; postFlag: progress to publish.
// ---------------------------------------------------------------------------
template <int KIN>
__device__ __forceinline__ void lstm_layer_body(
    const _Float16* __restrict__ xin,    // [B, T, KIN]
    const _Float16* __restrict__ Wcat,   // [1024, KIN+256]
    const float*    __restrict__ bsum,   // [1024]
    _Float16*       __restrict__ hseq,   // [B, T, 256] out
    int bblk, const int* waitFlag, int* postFlag) {
    constexpr int KCAT = KIN + H_;
    const int bBase = bblk * 16;
    const int tid   = threadIdx.x;
    const int wave  = tid >> 5;
    const int lane  = tid & 31;
    const int uBase = wave * 16;            // this wave's hidden slice

    __shared__ _Float16 hbuf[16][264];      // padded stride: conflict-free b128

    for (int i = tid; i < 16 * 264; i += 512)
        (&hbuf[0][0])[i] = (_Float16)0.0f;
    __syncthreads();

    // per-wave biases for the 4 gate column-tiles (q in i,f,g,o)
    float bq[4];
#pragma unroll
    for (int q = 0; q < 4; ++q)
        bq[q] = bsum[q * H_ + uBase + (lane & 15)];

    v8f cst;
#pragma unroll
    for (int r = 0; r < 8; ++r) cst[r] = 0.0f;

    for (int t = 0; t < T_; ++t) {
        // ---- pipeline wait: producer must have finished step t ----
        if (waitFlag) {
            if (tid == 0) {
                while (__hip_atomic_load(waitFlag, __ATOMIC_ACQUIRE,
                                         __HIP_MEMORY_SCOPE_AGENT) <= t) {
                    __builtin_amdgcn_s_sleep(1);
                }
            }
            __syncthreads();
        }

        // Block LICM from hoisting the (loop-invariant) weight-fragment loads
        // out of the 512-iteration loop and spilling them to scratch: this
        // may-write clobber pins all loads inside the iteration.
        asm volatile("" ::: "memory");

        // accumulators pre-seeded with the gate bias
        v8f acc[4];
#pragma unroll
        for (int q = 0; q < 4; ++q)
#pragma unroll
            for (int r = 0; r < 8; ++r) acc[q][r] = bq[q];

        // ---- K over the input x_t (global f16) ----
#pragma unroll
        for (int ki = 0; ki < KIN / 32; ++ki) {
            const int kk = ki * 32;
            v16h a = load_frag(xin + (size_t)t * KIN, bBase, T_ * KIN, kk, lane);
#pragma unroll
            for (int q = 0; q < 4; ++q) {
                v16h b = load_frag(Wcat, q * H_ + uBase, KCAT, kk, lane);
                acc[q] = __builtin_amdgcn_wmma_f32_16x16x32_f16(
                    false, a, false, b, (short)0, acc[q], false, false);
            }
        }
        // ---- K over h (LDS) ----
#pragma unroll
        for (int ki = 0; ki < H_ / 32; ++ki) {
            const int kk = ki * 32;
            v16h a = load_frag(&hbuf[0][0], 0, 264, kk, lane);
#pragma unroll
            for (int q = 0; q < 4; ++q) {
                v16h b = load_frag(Wcat, q * H_ + uBase, KCAT, KIN + kk, lane);
                acc[q] = __builtin_amdgcn_wmma_f32_16x16x32_f16(
                    false, a, false, b, (short)0, acc[q], false, false);
            }
        }

        // ---- LSTM cell update (wave-local, c stays in VGPRs) ----
        float hv[8];
#pragma unroll
        for (int r = 0; r < 8; ++r) {
            float iv = sigf(acc[0][r]);
            float fv = sigf(acc[1][r]);
            float gv = tanhf(acc[2][r]);
            float ov = sigf(acc[3][r]);
            float cv = fv * cst[r] + iv * gv;
            cst[r] = cv;
            hv[r]  = ov * tanhf(cv);
        }

        __syncthreads();   // all waves finished reading hbuf for this step
        {
            const int n    = lane & 15;
            const int rOff = (lane >> 4) * 8;
#pragma unroll
            for (int r = 0; r < 8; ++r)
                hbuf[rOff + r][uBase + n] = (_Float16)hv[r];
        }
        __syncthreads();   // h tile fully updated

        // cooperative copy of the new h tile to global hseq[b, t, :]
        {
            const int row = tid >> 5;          // 16 rows, 32 lanes per row
            const int cb  = lane * 8;          // 8 halves (16B) per lane
            *(uint4*)(hseq + ((size_t)(bBase + row) * T_ + t) * H_ + cb) =
                *(const uint4*)&hbuf[row][cb];
        }

        // ---- pipeline publish: step t visible to consumer ----
        if (postFlag) {
            __syncthreads();
            if (tid == 0) {
                __threadfence();
                __hip_atomic_store(postFlag, t + 1, __ATOMIC_RELEASE,
                                   __HIP_MEMORY_SCOPE_AGENT);
            }
        }
    }
}

// Persistent 2-layer pipeline: blocks 0..15 = layer 0, 16..31 = layer 1.
__global__ __launch_bounds__(512) void lstm2_kernel(
    const _Float16* __restrict__ xf16,
    _Float16* __restrict__ hseq0,
    _Float16* __restrict__ hseq1,
    const _Float16* __restrict__ Wcat0,
    const _Float16* __restrict__ Wcat1,
    const float* __restrict__ bsum0,
    const float* __restrict__ bsum1,
    int* __restrict__ flags) {
    const int layer = blockIdx.x >> 4;
    const int bblk  = blockIdx.x & 15;
    if (layer == 0) {
        lstm_layer_body<I_>(xf16, Wcat0, bsum0, hseq0, bblk,
                            nullptr, flags + bblk);
    } else {
        lstm_layer_body<H_>(hseq0, Wcat1, bsum1, hseq1, bblk,
                            flags + bblk, nullptr);
    }
}

// ---------------------------------------------------------------------------
// Classifier on last timestep: out[256,1000] = h_T @ fcWp^T + fcb
// ---------------------------------------------------------------------------
__global__ __launch_bounds__(32) void fc_kernel(
    const _Float16* __restrict__ hseq1,   // [B, T, H]
    const _Float16* __restrict__ fcWp,    // [1008, 256] f16, zero padded
    const float*    __restrict__ fcb,
    float*          __restrict__ out) {
    const int mBase = blockIdx.x * 16;
    const int nBase = blockIdx.y * 16;
    const int lane  = threadIdx.x;

    v8f acc;
#pragma unroll
    for (int r = 0; r < 8; ++r) acc[r] = 0.0f;

#pragma unroll
    for (int ki = 0; ki < H_ / 32; ++ki) {
        const int kk = ki * 32;
        v16h a = load_frag(hseq1 + (size_t)(T_ - 1) * H_, mBase, T_ * H_, kk, lane);
        v16h b = load_frag(fcWp, nBase, H_, kk, lane);
        acc = __builtin_amdgcn_wmma_f32_16x16x32_f16(
            false, a, false, b, (short)0, acc, false, false);
    }

    const int n   = lane & 15;
    const int col = nBase + n;
    if (col < C_) {
        const float bias = fcb[col];
        const int rOff = (lane >> 4) * 8;
#pragma unroll
        for (int r = 0; r < 8; ++r)
            out[(size_t)(mBase + rOff + r) * C_ + col] = acc[r] + bias;
    }
}

// ---------------------------------------------------------------------------
extern "C" void kernel_launch(void* const* d_in, const int* in_sizes, int n_in,
                              void* d_out, int out_size, void* d_ws, size_t ws_size,
                              hipStream_t stream) {
    (void)in_sizes; (void)n_in; (void)out_size; (void)ws_size;

    const float* x    = (const float*)d_in[0];
    const float* Wih0 = (const float*)d_in[1];
    const float* Whh0 = (const float*)d_in[2];
    const float* bih0 = (const float*)d_in[3];
    const float* bhh0 = (const float*)d_in[4];
    const float* Wih1 = (const float*)d_in[5];
    const float* Whh1 = (const float*)d_in[6];
    const float* bih1 = (const float*)d_in[7];
    const float* bhh1 = (const float*)d_in[8];
    const float* fcW  = (const float*)d_in[9];
    const float* fcb  = (const float*)d_in[10];
    float* out = (float*)d_out;

    // workspace carve-up (all 256B aligned by construction)
    char* ws = (char*)d_ws;
    size_t off = 0;
    _Float16* xf16  = (_Float16*)(ws + off); off += (size_t)B_ * T_ * I_ * 2;   // 32 MB
    _Float16* hseq0 = (_Float16*)(ws + off); off += (size_t)B_ * T_ * H_ * 2;   // 64 MB
    _Float16* hseq1 = (_Float16*)(ws + off); off += (size_t)B_ * T_ * H_ * 2;   // 64 MB
    _Float16* Wcat0 = (_Float16*)(ws + off); off += (size_t)G_ * (I_ + H_) * 2;
    _Float16* Wcat1 = (_Float16*)(ws + off); off += (size_t)G_ * (H_ + H_) * 2;
    float*    bsum0 = (float*)   (ws + off); off += (size_t)G_ * 4;
    float*    bsum1 = (float*)   (ws + off); off += (size_t)G_ * 4;
    _Float16* fcWp  = (_Float16*)(ws + off); off += (size_t)CPAD_ * H_ * 2;
    int*      flags = (int*)     (ws + off); off += 256;

    // ---- prep: flag reset, precision conversion, weight packing ----
    zero_flags<<<1, 32, 0, stream>>>(flags, 16);
    {
        int n = B_ * T_ * I_;
        cvt_f32_to_f16<<<(n + 255) / 256, 256, 0, stream>>>(x, xf16, n);
    }
    {
        int n = G_ * (I_ + H_);
        build_wcat<<<(n + 255) / 256, 256, 0, stream>>>(Wih0, Whh0, Wcat0, I_);
    }
    {
        int n = G_ * (H_ + H_);
        build_wcat<<<(n + 255) / 256, 256, 0, stream>>>(Wih1, Whh1, Wcat1, H_);
    }
    build_bsum<<<(G_ + 255) / 256, 256, 0, stream>>>(bih0, bhh0, bsum0);
    build_bsum<<<(G_ + 255) / 256, 256, 0, stream>>>(bih1, bhh1, bsum1);
    {
        int n = CPAD_ * H_;
        build_fcw<<<(n + 255) / 256, 256, 0, stream>>>(fcW, fcWp);
    }

    // ---- pipelined 2-layer recurrence: 32 persistent WGs x 16 waves ----
    lstm2_kernel<<<dim3(32), 512, 0, stream>>>(xf16, hseq0, hseq1,
                                               Wcat0, Wcat1, bsum0, bsum1, flags);

    // ---- classifier ----
    fc_kernel<<<dim3(B_ / 16, CPAD_ / 16), 32, 0, stream>>>(hseq1, fcWp, fcb, out);
}
